// _YatPerformerPolyBase_56367150793402
// MI455X (gfx1250) — matrix operand order repaired and checked by reading, not compiled, not run on hardware
//
#include <hip/hip_runtime.h>
#include <hip/hip_bf16.h>
#include <math.h>

// ---------------------------------------------------------------------------
// Problem dimensions (fixed by the reference setup)
// ---------------------------------------------------------------------------
constexpr int T  = 1024;      // tokens
constexpr int E  = 512;       // embed
constexpr int E3 = 3 * E;     // qkv rows
constexpr int H  = 8;         // heads
constexpr int D  = 64;        // head dim
constexpr int MM = 8;         // random-feature dim M
constexpr int P  = 32;        // poly dim
constexpr int F  = P * MM;    // feature dim = 256
constexpr int CH = 128;       // chunk
constexpr int NC = T / CH;    // 8 chunks

typedef __attribute__((ext_vector_type(16))) __bf16 v16bf;
typedef __attribute__((ext_vector_type(8)))  float  v8f;

#if __has_builtin(__builtin_amdgcn_tensor_load_to_lds)
#define USE_TDM 1
#else
#define USE_TDM 0
#endif

// ---------------------------------------------------------------------------
// WMMA wrapper: D(f32 16x16) = A(bf16 16x32) * B(bf16 32x16) + C
// ---------------------------------------------------------------------------
__device__ inline v8f wmma_bf16(v16bf a, v16bf b, v8f c) {
  return __builtin_amdgcn_wmma_f32_16x16x32_bf16(
      /*neg_a=*/false, a, /*neg_b=*/false, b,
      /*c_mod=*/(short)0, c, /*reuse_a=*/false, /*reuse_b=*/false);
}

// ---------------------------------------------------------------------------
// Fragment loaders (wave32), contiguous k order used for both A and B.
// A 16x32 bf16: lane holds row m=lane&15; element j -> k = j + (hi?16:0)
// B 32x16 bf16: lane holds col n=lane&15; element j -> k = j + (hi?16:0)
// C/D 16x16 f32: vgpr v -> m = v + (hi?8:0), n = lane&15
// ---------------------------------------------------------------------------
__device__ inline v16bf load_frag_a(const __bf16* base, int m0, int ld, int k0) {
  int lane = threadIdx.x & 31;
  v16bf a;
#pragma unroll
  for (int j = 0; j < 16; ++j) {
    int m = lane & 15;
    int k = j + ((lane & 16) ? 16 : 0);
    a[j] = base[(m0 + m) * ld + k0 + k];
  }
  return a;
}
// logical A[m][k] = base[k*ld + m]  (transposed source, e.g. Kf^T)
__device__ inline v16bf load_frag_a_t(const __bf16* base, int m0, int ld, int k0) {
  int lane = threadIdx.x & 31;
  v16bf a;
#pragma unroll
  for (int j = 0; j < 16; ++j) {
    int m = lane & 15;
    int k = j + ((lane & 16) ? 16 : 0);
    a[j] = base[(k0 + k) * ld + m0 + m];
  }
  return a;
}
// logical B[k][n] = base[k*ld + n]
__device__ inline v16bf load_frag_b(const __bf16* base, int k0, int ld, int n0) {
  int lane = threadIdx.x & 31;
  v16bf b;
#pragma unroll
  for (int j = 0; j < 16; ++j) {
    int n = lane & 15;
    int k = j + ((lane & 16) ? 16 : 0);
    b[j] = base[(k0 + k) * ld + n0 + n];
  }
  return b;
}
// logical B[k][n] = base[n*ld + k]  (row-major weight W[n][k], or Kf[s][f])
__device__ inline v16bf load_frag_b_t(const __bf16* base, int k0, int ld, int n0) {
  int lane = threadIdx.x & 31;
  v16bf b;
#pragma unroll
  for (int j = 0; j < 16; ++j) {
    int n = lane & 15;
    int k = j + ((lane & 16) ? 16 : 0);
    b[j] = base[(n0 + n) * ld + k0 + k];
  }
  return b;
}
// B fragment converted on the fly from an f32 source (KV state in LDS)
__device__ inline v16bf load_frag_b_f32(const float* base, int k0, int ld, int n0) {
  int lane = threadIdx.x & 31;
  v16bf b;
#pragma unroll
  for (int j = 0; j < 16; ++j) {
    int n = lane & 15;
    int k = j + ((lane & 16) ? 16 : 0);
    b[j] = (__bf16)base[(k0 + k) * ld + n0 + n];
  }
  return b;
}
__device__ inline v8f load_c_f32(const float* base, int m0, int ld, int n0) {
  int lane = threadIdx.x & 31;
  v8f c;
#pragma unroll
  for (int v = 0; v < 8; ++v) {
    int m = v + ((lane & 16) ? 8 : 0);
    int n = lane & 15;
    c[v] = base[(m0 + m) * ld + n0 + n];
  }
  return c;
}
__device__ inline void store_c_f32(float* base, int m0, int ld, int n0, v8f c) {
  int lane = threadIdx.x & 31;
#pragma unroll
  for (int v = 0; v < 8; ++v) {
    int m = v + ((lane & 16) ? 8 : 0);
    int n = lane & 15;
    base[(m0 + m) * ld + n0 + n] = c[v];
  }
}

// ---------------------------------------------------------------------------
// Tensor Data Mover descriptors (CDNA5 ISA D# layout).
// Group0: [1:0]=count, [63:32]=lds_addr, [120:64]=global_addr, [127:126]=type=2
// Group1: [17:16]=data_size(1->2B), [79:48]=tensor_dim0, [111:80]=tensor_dim1,
//         [127:112]=tile_dim0, [143:128]=tile_dim1, [207:160]=tensor_dim0_stride
// ---------------------------------------------------------------------------
#if USE_TDM
typedef unsigned int v4u __attribute__((ext_vector_type(4)));
typedef int v4i __attribute__((ext_vector_type(4)));
typedef int v8i __attribute__((ext_vector_type(8)));

__device__ inline unsigned int lds_off(void* p) {
  return (unsigned int)(uintptr_t)(__attribute__((address_space(3))) void*)p;
}

__device__ inline void tdm_desc(v4u& g0, v8i& g1, unsigned int lds,
                                const void* gptr, unsigned int tile0,
                                unsigned int tile1, unsigned int tdim0,
                                unsigned int tdim1, unsigned int stride0) {
  unsigned long long ga = (unsigned long long)(uintptr_t)gptr;
  g0[0] = 1u;                                   // count = 1 valid descriptor
  g0[1] = lds;                                  // lds_addr (bytes)
  g0[2] = (unsigned int)ga;                     // global_addr[31:0]
  g0[3] = (unsigned int)((ga >> 32) & 0x01ffffffu) | (2u << 30);  // addr hi, type=2
  g1[0] = 0x00010000;                           // data_size = 1 (2 bytes/elem)
  g1[1] = (int)((tdim0 & 0xffffu) << 16);       // tensor_dim0[15:0] at bits 63:48
  g1[2] = (int)(((tdim0 >> 16) & 0xffffu) | ((tdim1 & 0xffffu) << 16));
  g1[3] = (int)(((tdim1 >> 16) & 0xffffu) | ((tile0 & 0xffffu) << 16));
  g1[4] = (int)(tile1 & 0xffffu);               // tile_dim1; tile_dim2 = 0
  g1[5] = (int)stride0;                         // tensor_dim0_stride[31:0]
  g1[6] = 0;                                    // stride0 hi, stride1 lo = 0
  g1[7] = 0;
}

__device__ inline void tdm_load(unsigned int lds, const void* gptr,
                                unsigned int tile0, unsigned int tile1,
                                unsigned int tdim0, unsigned int tdim1,
                                unsigned int stride0) {
  v4u g0; v8i g1;
  tdm_desc(g0, g1, lds, gptr, tile0, tile1, tdim0, tdim1, stride0);
  v4i gz = {0, 0, 0, 0};
#if __clang_major__ >= 23
  v8i gz8 = {0, 0, 0, 0, 0, 0, 0, 0};
  __builtin_amdgcn_tensor_load_to_lds(g0, g1, gz, gz, gz8, 0);
#else
  __builtin_amdgcn_tensor_load_to_lds(g0, g1, gz, gz, 0);
#endif
}

__device__ inline void tdm_store(unsigned int lds, void* gptr,
                                 unsigned int tile0, unsigned int tile1,
                                 unsigned int tdim0, unsigned int tdim1,
                                 unsigned int stride0) {
  v4u g0; v8i g1;
  tdm_desc(g0, g1, lds, gptr, tile0, tile1, tdim0, tdim1, stride0);
  v4i gz = {0, 0, 0, 0};
#if __clang_major__ >= 23
  v8i gz8 = {0, 0, 0, 0, 0, 0, 0, 0};
  __builtin_amdgcn_tensor_store_from_lds(g0, g1, gz, gz, gz8, 0);
#else
  __builtin_amdgcn_tensor_store_from_lds(g0, g1, gz, gz, 0);
#endif
}
#endif  // USE_TDM

// ---------------------------------------------------------------------------
// K0: f32 -> bf16 conversion
// ---------------------------------------------------------------------------
__global__ void k_cvt(const float* __restrict__ s, __bf16* __restrict__ d, int n) {
  int i = blockIdx.x * blockDim.x + threadIdx.x;
  if (i < n) d[i] = (__bf16)s[i];
}

// ---------------------------------------------------------------------------
// K1: qkv = x @ W_qkv^T + b_qkv. q,k -> f32 [2][H][T][D]; v -> bf16 [H][T][D].
// One 16x16 output tile per wave, WMMA bf16 over K=E.
// ---------------------------------------------------------------------------
__global__ void k_qkv(const __bf16* __restrict__ xb, const __bf16* __restrict__ wb,
                      const float* __restrict__ bias, float* __restrict__ qk,
                      __bf16* __restrict__ vb) {
  int wid  = threadIdx.x >> 5;
  int lane = threadIdx.x & 31;
  int tile = blockIdx.x * 8 + wid;
  const int NT = E3 / 16;            // 96 tiles along N
  int tm = tile / NT, tn = tile % NT;
  v8f c = {};
  for (int k0 = 0; k0 < E; k0 += 32) {
    v16bf a = load_frag_a(xb, tm * 16, E, k0);
    v16bf b = load_frag_b_t(wb, k0, E, tn * 16);   // W stored [N=E3][K=E]
    c = wmma_bf16(a, b, c);
  }
#pragma unroll
  for (int v = 0; v < 8; ++v) {
    int m = v + ((lane & 16) ? 8 : 0);
    int n = lane & 15;
    int t = tm * 16 + m;
    int e = tn * 16 + n;
    float val = c[v] + bias[e];
    int which = e >> 9;              // 0:q 1:k 2:v
    int rem   = e & 511;
    int h = rem >> 6, d = rem & 63;
    if (which == 2)
      vb[((size_t)h * T + t) * D + d] = (__bf16)val;
    else
      qk[(((size_t)which * H + h) * T + t) * D + d] = val;
  }
}

// ---------------------------------------------------------------------------
// K2: feature map. One thread per (which, h, t):
//   xn = x/||x||; poly = xn@W_poly[h]; proj = xn@omega[0,h];
//   prf = exp(clip(proj*sqrt(2s)-s,-20,20))/sqrt(M)*sqrt(qw);
//   feat[p*M+m] = poly[p]*prf[m]  -> bf16 [2][H][T][F]
// ---------------------------------------------------------------------------
__global__ void k_feat(const float* __restrict__ qk, const float* __restrict__ W_poly,
                       const float* __restrict__ omega, const float* __restrict__ qn,
                       const float* __restrict__ qw, __bf16* __restrict__ feat) {
  int gid = blockIdx.x * blockDim.x + threadIdx.x;
  if (gid >= 2 * H * T) return;
  int which = gid / (H * T);
  int rem   = gid % (H * T);
  int h = rem / T, t = rem % T;
  const float* xc = qk + (((size_t)which * H + h) * T + t) * D;

  float ss = 0.f;
  for (int d = 0; d < D; ++d) { float x = xc[d]; ss += x * x; }
  float inv = 1.0f / fmaxf(sqrtf(ss), 1e-12f);

  float poly[P] = {};
  float proj[MM] = {};
  const float* Wp = W_poly + (size_t)h * D * P;
  const float* om = omega + (size_t)h * D * MM;
  for (int d = 0; d < D; ++d) {
    float xv = xc[d] * inv;
#pragma unroll
    for (int p = 0; p < P; ++p) poly[p] += xv * Wp[d * P + p];
#pragma unroll
    for (int m = 0; m < MM; ++m) proj[m] += xv * om[d * MM + m];
  }
  float s     = qn[0];
  float sq2s  = sqrtf(2.0f * fmaxf(s, 0.0f));
  float scale = sqrtf(fmaxf(qw[0], 0.0f)) * 0.35355339059327373f; // sqrt(qw)/sqrt(M)
  float prf[MM];
#pragma unroll
  for (int m = 0; m < MM; ++m)
    prf[m] = expf(fminf(fmaxf(proj[m] * sq2s - s, -20.0f), 20.0f)) * scale;

  __bf16* o = feat + (((size_t)which * H + h) * T + t) * F;
#pragma unroll
  for (int p = 0; p < P; ++p)
#pragma unroll
    for (int m = 0; m < MM; ++m)
      o[p * MM + m] = (__bf16)(poly[p] * prf[m]);
}

// ---------------------------------------------------------------------------
// K3: chunked linear attention. One block (16 waves) per head; chunks serial.
// Chunk tiles & recurrent state live in LDS (~258 KB of the 320 KB/WGP).
// TDM moves Qf/Kf/V chunks into LDS and O tiles back out (TENSORcnt-tracked);
// the SIMDs only run WMMA:
//   S  = Qf Kf^T (masked)           -> 288 WMMA / chunk
//   O  = S V + Qf KV_state          -> 384 WMMA
//   KV_state += Kf^T V              -> 256 WMMA
// ---------------------------------------------------------------------------
__global__ void __launch_bounds__(512)
k_attn(const __bf16* __restrict__ Qf, const __bf16* __restrict__ Kf,
       const __bf16* __restrict__ Vg, __bf16* __restrict__ attn) {
  __shared__ __bf16 sQ[CH * F];    // 64 KB
  __shared__ __bf16 sK[CH * F];    // 64 KB
  __shared__ __bf16 sV[CH * D];    // 16 KB
  __shared__ __bf16 sS[CH * CH];   // 32 KB
  __shared__ __bf16 sO[CH * D];    // 16 KB (staged output tile)
  __shared__ float  sKV[F * D];    // 64 KB
  __shared__ float  sKst[F];
  __shared__ float  sDen1[CH];
  __shared__ float  sDen2[CH];

  int h = blockIdx.x;
  int tid = threadIdx.x;
  int wid = tid >> 5, lane = tid & 31;

  for (int i = tid; i < F * D; i += 512) sKV[i] = 0.f;
  for (int i = tid; i < F; i += 512) sKst[i] = 0.f;
  __syncthreads();

  for (int c = 0; c < NC; ++c) {
    int tbase = c * CH;
    const __bf16* Qg = Qf + ((size_t)h * T + tbase) * F;
    const __bf16* Kg = Kf + ((size_t)h * T + tbase) * F;
    const __bf16* Vp = Vg + ((size_t)h * T + tbase) * D;
#if USE_TDM
    if (wid == 0) {   // wave-uniform branch: only wave 0 issues the DMAs
      tdm_load(lds_off(sQ), Qg, CH * F, 1, CH * F, 1, CH * F);
      tdm_load(lds_off(sK), Kg, CH * F, 1, CH * F, 1, CH * F);
      tdm_load(lds_off(sV), Vp, CH * D, 1, CH * D, 1, CH * D);
      __builtin_amdgcn_s_wait_tensorcnt((short)0);  // loads + prior O store done
    }
#else
    for (int i = tid; i < CH * F; i += 512) { sQ[i] = Qg[i]; sK[i] = Kg[i]; }
    for (int i = tid; i < CH * D; i += 512) sV[i] = Vp[i];
#endif
    __syncthreads();

    // den2[t] = Qf[t] . k_state   (old state)
    for (int t = tid; t < CH; t += 512) {
      float acc = 0.f;
      for (int f = 0; f < F; ++f) acc += (float)sQ[t * F + f] * sKst[f];
      sDen2[t] = acc;
    }

    // S = mask(Qf Kf^T): 64 tiles over 16 waves
    for (int tile = wid; tile < 64; tile += 16) {
      int ti = tile >> 3, tj = tile & 7;
      if (tj > ti) {  // strictly-upper tile: zero
#pragma unroll
        for (int v = 0; v < 8; ++v) {
          int m = v + ((lane & 16) ? 8 : 0), n = lane & 15;
          sS[(ti * 16 + m) * CH + tj * 16 + n] = (__bf16)0.f;
        }
        continue;
      }
      v8f acc = {};
      for (int k0 = 0; k0 < F; k0 += 32) {
        v16bf a = load_frag_a(sQ, ti * 16, F, k0);
        v16bf b = load_frag_b_t(sK, k0, F, tj * 16);   // B[k=f][n=s] = Kf[s][f]
        acc = wmma_bf16(a, b, acc);
      }
#pragma unroll
      for (int v = 0; v < 8; ++v) {
        int m = v + ((lane & 16) ? 8 : 0), n = lane & 15;
        int tg = ti * 16 + m, sg = tj * 16 + n;
        sS[tg * CH + sg] = (__bf16)((sg <= tg) ? acc[v] : 0.f);
      }
    }
    __syncthreads();

    // den1[t] = rowsum(mask(S))
    for (int t = tid; t < CH; t += 512) {
      float acc = 0.f;
      for (int s = 0; s < CH; ++s) acc += (float)sS[t * CH + s];
      sDen1[t] = acc;
    }
    __syncthreads();

    // O = S V + Qf KV_state, then /den : 32 tiles over 16 waves
    for (int tile = wid; tile < 32; tile += 16) {
      int ti = tile >> 2, dj = tile & 3;
      v8f acc = {};
      for (int k0 = 0; k0 < CH; k0 += 32) {
        v16bf a = load_frag_a(sS, ti * 16, CH, k0);
        v16bf b = load_frag_b(sV, k0, D, dj * 16);
        acc = wmma_bf16(a, b, acc);
      }
      for (int k0 = 0; k0 < F; k0 += 32) {
        v16bf a = load_frag_a(sQ, ti * 16, F, k0);
        v16bf b = load_frag_b_f32(sKV, k0, D, dj * 16);
        acc = wmma_bf16(a, b, acc);
      }
#pragma unroll
      for (int v = 0; v < 8; ++v) {
        int m = v + ((lane & 16) ? 8 : 0), n = lane & 15;
        int tl = ti * 16 + m, d = dj * 16 + n;
        float den = fmaxf(sDen1[tl] + sDen2[tl], 1e-6f);
#if USE_TDM
        sO[tl * D + d] = (__bf16)(acc[v] / den);
#else
        attn[(size_t)(tbase + tl) * E + h * D + d] = (__bf16)(acc[v] / den);
#endif
      }
    }
    __syncthreads();

#if USE_TDM
    // 2-D TDM store: 128 rows x 64 elems, global row stride E=512 elems
    if (wid == 0)
      tdm_store(lds_off(sO), attn + ((size_t)tbase * E + h * D),
                D, CH, D, CH, E);
#endif

    // KV_state += Kf^T V : 64 tiles over 16 waves (RMW of own tile)
    for (int tile = wid; tile < 64; tile += 16) {
      int fi = tile >> 2, dj = tile & 3;
      v8f acc = load_c_f32(sKV, fi * 16, D, dj * 16);
      for (int k0 = 0; k0 < CH; k0 += 32) {
        v16bf a = load_frag_a_t(sK, fi * 16, F, k0);  // A[m=f][k=s] = Kf[s][f]
        v16bf b = load_frag_b(sV, k0, D, dj * 16);
        acc = wmma_bf16(a, b, acc);
      }
      store_c_f32(sKV, fi * 16, D, dj * 16, acc);
    }
    // k_state += column sums of Kf
    for (int f = tid; f < F; f += 512) {
      float acc = 0.f;
      for (int s = 0; s < CH; ++s) acc += (float)sK[s * F + f];
      sKst[f] += acc;
    }
    __syncthreads();
  }
  // s_endpgm performs an implicit wait-idle, covering the final TDM store.
}

// ---------------------------------------------------------------------------
// K4: y = attn @ W_out^T + b_out  (f32 out to d_out)
// ---------------------------------------------------------------------------
__global__ void k_outproj(const __bf16* __restrict__ attn, const __bf16* __restrict__ wo,
                          const float* __restrict__ bias, float* __restrict__ y) {
  int wid  = threadIdx.x >> 5;
  int lane = threadIdx.x & 31;
  int tile = blockIdx.x * 8 + wid;
  const int NT = E / 16;             // 32 tiles along N
  int tm = tile / NT, tn = tile % NT;
  v8f c = {};
  for (int k0 = 0; k0 < E; k0 += 32) {
    v16bf a = load_frag_a(attn, tm * 16, E, k0);
    v16bf b = load_frag_b_t(wo, k0, E, tn * 16);
    c = wmma_bf16(a, b, c);
  }
#pragma unroll
  for (int v = 0; v < 8; ++v) {
    int m = v + ((lane & 16) ? 8 : 0);
    int n = lane & 15;
    y[(size_t)(tm * 16 + m) * E + tn * 16 + n] = c[v] + bias[tn * 16 + n];
  }
}

// ---------------------------------------------------------------------------
// Host launcher
// ---------------------------------------------------------------------------
extern "C" void kernel_launch(void* const* d_in, const int* in_sizes, int n_in,
                              void* d_out, int out_size, void* d_ws, size_t ws_size,
                              hipStream_t stream) {
  const float* x      = (const float*)d_in[0];
  const float* W_qkv  = (const float*)d_in[1];
  const float* b_qkv  = (const float*)d_in[2];
  const float* W_out  = (const float*)d_in[3];
  const float* b_out  = (const float*)d_in[4];
  const float* omega  = (const float*)d_in[5];
  const float* W_poly = (const float*)d_in[6];
  const float* qnode  = (const float*)d_in[7];
  const float* qweight= (const float*)d_in[8];
  float* y = (float*)d_out;

  char* ws = (char*)d_ws;
  size_t off = 0;
  auto take = [&](size_t bytes) -> char* {
    char* p = ws + off;
    off = (off + bytes + 255) & ~(size_t)255;
    return p;
  };
  __bf16* xb    = (__bf16*)take((size_t)T * E * 2);
  __bf16* wqb   = (__bf16*)take((size_t)E3 * E * 2);
  __bf16* wob   = (__bf16*)take((size_t)E * E * 2);
  float*  qkf   = (float*) take((size_t)2 * H * T * D * 4);   // q,k f32
  __bf16* vbf   = (__bf16*)take((size_t)H * T * D * 2);       // v bf16
  __bf16* featb = (__bf16*)take((size_t)2 * H * T * F * 2);   // [2][H][T][F]
  __bf16* attnb = (__bf16*)take((size_t)T * E * 2);

  // K0: conversions
  k_cvt<<<(T * E + 255) / 256, 256, 0, stream>>>(x, xb, T * E);
  k_cvt<<<(E3 * E + 255) / 256, 256, 0, stream>>>(W_qkv, wqb, E3 * E);
  k_cvt<<<(E * E + 255) / 256, 256, 0, stream>>>(W_out, wob, E * E);

  // K1: qkv projection (6144 tiles, 8 waves/block)
  k_qkv<<<(T / 16) * (E3 / 16) / 8, 256, 0, stream>>>(xb, wqb, b_qkv, qkf, vbf);

  // K2: feature maps for q and k
  k_feat<<<(2 * H * T + 255) / 256, 256, 0, stream>>>(qkf, W_poly, omega, qnode,
                                                      qweight, featb);

  // K3: chunked linear attention, one block per head
  const __bf16* Qf = featb;
  const __bf16* Kf = featb + (size_t)H * T * F;
  k_attn<<<H, 512, 0, stream>>>(Qf, Kf, vbf, attnb);

  // K4: output projection (2048 tiles)
  k_outproj<<<(T / 16) * (E / 16) / 8, 256, 0, stream>>>(attnb, wob, b_out, y);

  (void)in_sizes; (void)n_in; (void)out_size; (void)ws_size;
}
